// MinimalRNNCell_65859028517455
// MI455X (gfx1250) — matrix operand order repaired and checked by reading, not compiled, not run on hardware
//
#include <hip/hip_runtime.h>

// CDNA5 / gfx1250: wave32, v_wmma_f32_16x16x32_bf16,
// TDM tensor_load_to_lds for B tiles + async global->LDS for A tiles.

typedef __attribute__((ext_vector_type(16))) __bf16     v16bf;
typedef __attribute__((ext_vector_type(8)))  float      v8f;
typedef __attribute__((ext_vector_type(8)))  unsigned   v8u;
typedef __attribute__((ext_vector_type(4)))  unsigned   v4u;
typedef __attribute__((ext_vector_type(8)))  int        v8i;
typedef __attribute__((ext_vector_type(4)))  int        v4i;

#define TILE_M 32
#define TILE_N 256
#define KBLK   64

#if defined(__has_builtin)
#  if __has_builtin(__builtin_amdgcn_tensor_load_to_lds)
#    define HAVE_TDM_BUILTIN 1
#  endif
#  if __has_builtin(__builtin_amdgcn_s_wait_tensorcnt)
#    define HAVE_TENSORCNT_BUILTIN 1
#  endif
#endif

// ---- CDNA5 helpers --------------------------------------------------------

__device__ __forceinline__ unsigned lds_off(const void* p) {
    // generic -> LDS address space -> 32-bit LDS byte offset
    return (unsigned)(unsigned long long)(__attribute__((address_space(3))) const char*)p;
}

__device__ __forceinline__ void async_ld_b128(unsigned ldsAddr, const void* g) {
    unsigned long long ga = (unsigned long long)g;
    asm volatile("global_load_async_to_lds_b128 %0, %1, off"
                 :: "v"(ldsAddr), "v"(ga) : "memory");
}

__device__ __forceinline__ void wait_async_le1() {
    asm volatile("s_wait_asynccnt 0x1" ::: "memory");
}
__device__ __forceinline__ void wait_async_0() {
    asm volatile("s_wait_asynccnt 0x0" ::: "memory");
}

__device__ __forceinline__ void wait_tensor_le1() {
#ifdef HAVE_TENSORCNT_BUILTIN
    __builtin_amdgcn_s_wait_tensorcnt((short)1);
#else
    asm volatile("s_wait_tensorcnt 0x1" ::: "memory");
#endif
}
__device__ __forceinline__ void wait_tensor_0() {
#ifdef HAVE_TENSORCNT_BUILTIN
    __builtin_amdgcn_s_wait_tensorcnt((short)0);
#else
    asm volatile("s_wait_tensorcnt 0x0" ::: "memory");
#endif
}

// TDM 2D tile load: nRows rows of rowLen bf16 elements, row stride rowStride
// (elements), packed contiguously into LDS at ldsAddr. D# per ISA 8.3/8.4.
__device__ __forceinline__ void tdm_load_tile_2d(unsigned ldsAddr, const void* gtile,
                                                 int rowLen, int nRows, int rowStride,
                                                 int tDim0, int tDim1)
{
    unsigned long long ga = (unsigned long long)gtile;
    v4u g0 = { 1u,                                   // count=1, user mode
               ldsAddr,                              // lds_addr
               (unsigned)ga,                         // global_addr[31:0]
               (unsigned)((ga >> 32) & 0x01FFFFFFu) | 0x80000000u };  // [56:32] | type=2
    v8i g1 = { (int)0x00010000u,                     // data_size=1 (2 bytes), no extras
               (int)(((unsigned)tDim0 & 0xFFFFu) << 16),                   // tensor_dim0 lo16
               (int)((((unsigned)tDim0 >> 16) & 0xFFFFu)
                     | (((unsigned)tDim1 & 0xFFFFu) << 16)),               // dim0 hi / dim1 lo
               (int)((((unsigned)tDim1 >> 16) & 0xFFFFu)
                     | (((unsigned)rowLen & 0xFFFFu) << 16)),              // dim1 hi / tile_dim0
               (int)((unsigned)nRows & 0xFFFFu),     // tile_dim1 (tile_dim2 = 0)
               rowStride,                            // tensor_dim0_stride[31:0]
               0, 0 };
    v4i g2 = {0, 0, 0, 0};
    v4i g3 = {0, 0, 0, 0};
#ifdef HAVE_TDM_BUILTIN
#  if __clang_major__ >= 23
    v8i g4 = {0, 0, 0, 0, 0, 0, 0, 0};
    __builtin_amdgcn_tensor_load_to_lds(g0, g1, g2, g3, g4, 0);
#  else
    __builtin_amdgcn_tensor_load_to_lds(g0, g1, g2, g3, 0);
#  endif
#else
    asm volatile("tensor_load_to_lds %0, %1, %2, %3"
                 :: "s"(g0), "s"(g1), "s"(g2), "s"(g3) : "memory");
#endif
}

// ---- one-shot converts ----------------------------------------------------

__global__ __launch_bounds__(256)
void cvt_f32_bf16(const float* __restrict__ in, __bf16* __restrict__ out, long n) {
    long i = ((long)blockIdx.x * blockDim.x + threadIdx.x) * 8;
    if (i + 8 > n) return;
    float4 a = *(const float4*)(in + i);
    float4 b = *(const float4*)(in + i + 4);
    __bf16 t[8] = {(__bf16)a.x, (__bf16)a.y, (__bf16)a.z, (__bf16)a.w,
                   (__bf16)b.x, (__bf16)b.y, (__bf16)b.z, (__bf16)b.w};
    *(uint4*)(out + i) = *(const uint4*)t;
}

__global__ __launch_bounds__(256)
void cvt_f32_bf16_T(const float* __restrict__ in, __bf16* __restrict__ outT,
                    int R, int C) {   // in[R][C] -> outT[C][R]
    long i = (long)blockIdx.x * blockDim.x + threadIdx.x;
    if (i >= (long)R * C) return;
    int r = (int)(i / C), c = (int)(i % C);
    outT[(long)c * R + r] = (__bf16)in[i];
}

// ---- generic bf16 WMMA GEMM ----------------------------------------------
// Out[b,m,n] = sum_k A[b,m,k] * Bt[n,k]  (+ Cadd[b,m,n])
// A: bf16 row-major (strides in elems). Bt: bf16 [N][K] (B pre-transposed).
// Output: bf16 (OutB) or f32 (OutF); OutT optionally also stores Out^T (bf16).
// Grid: x = N/256, y = M/32, z = batches. Block = 256 threads = 8 wave32.
__global__ __launch_bounds__(256)
void minrnn_gemm(const __bf16* __restrict__ A, long aRow, long aBatch,
                 const __bf16* __restrict__ Bt,
                 const __bf16* __restrict__ Cadd, long cRow, long cBatch,
                 __bf16* __restrict__ OutB, float* __restrict__ OutF,
                 long oRow, long oBatch,
                 __bf16* __restrict__ OutT, int Mtot,
                 int N, int K)
{
    __shared__ __bf16 Albs[2][TILE_M][KBLK] __attribute__((aligned(32)));  // 2 x 4 KB
    __shared__ __bf16 Blds[2][TILE_N][KBLK] __attribute__((aligned(32)));  // 2 x 32 KB

    const int tid  = threadIdx.x;
    const int lane = tid & 31;
    const int wave = tid >> 5;
    const int batch = blockIdx.z;
    const int nBase = blockIdx.x * TILE_N;
    const int mBase = blockIdx.y * TILE_M;

    const int halfSel = lane >> 4;       // 0: lanes 0-15, 1: lanes 16-31
    const int l15     = lane & 15;
    const int aKlo    = halfSel ? 16 : 0; // A frag: K 0..15 / 16..31
    const int bKoff   = halfSel ? 16 : 0;// B frag: K 0..15 / 16..31

    const __bf16* Ab = A + (long)batch * aBatch + (long)mBase * aRow;
    const __bf16* Bb = Bt + (long)nBase * K;     // 256-row tile base

    // A staging: 32x64 tile, one async b128 per thread
    const int sar = tid >> 3;            // A row 0..31
    const int sac = (tid & 7) << 3;      // A col 0,8,..,56

    v8f acc[2][2] = {};
    const int nStages = K / KBLK;

    // prologue: stage buffer 0
    async_ld_b128(lds_off(&Albs[0][sar][sac]), Ab + (long)sar * aRow + sac);
    if (wave == 0)
        tdm_load_tile_2d(lds_off(&Blds[0][0][0]), Bb, KBLK, TILE_N, K, K, N);

    for (int s = 0; s < nStages; ++s) {
        const int  cur = s & 1;
        const bool hasNext = (s + 1 < nStages);
        if (hasNext) {
            const int nk = (s + 1) * KBLK;
            async_ld_b128(lds_off(&Albs[cur ^ 1][sar][sac]),
                          Ab + (long)sar * aRow + nk + sac);
            if (wave == 0)
                tdm_load_tile_2d(lds_off(&Blds[cur ^ 1][0][0]), Bb + nk,
                                 KBLK, TILE_N, K, K, N);
        }
        // in-order completion per counter: <=1 outstanding -> current stage landed
        if (hasNext) { wait_async_le1(); wait_tensor_le1(); }
        else         { wait_async_0();  wait_tensor_0();  }
        __syncthreads();

        #pragma unroll
        for (int kb = 0; kb < KBLK; kb += 32) {
            v16bf bfrag[2], afrag[2];
            #pragma unroll
            for (int nt = 0; nt < 2; ++nt)
                bfrag[nt] = *(const v16bf*)&Blds[cur][wave * 32 + nt * 16 + l15][kb + bKoff];
            #pragma unroll
            for (int mt = 0; mt < 2; ++mt) {
                const __bf16* ar = &Albs[cur][mt * 16 + l15][kb];
                uint4 u0 = *(const uint4*)(ar + aKlo);
                uint4 u1 = *(const uint4*)(ar + aKlo + 8);
                v8u aw = {u0.x, u0.y, u0.z, u0.w, u1.x, u1.y, u1.z, u1.w};
                afrag[mt] = __builtin_bit_cast(v16bf, aw);
            }
            #pragma unroll
            for (int mt = 0; mt < 2; ++mt)
                #pragma unroll
                for (int nt = 0; nt < 2; ++nt)
                    acc[mt][nt] = __builtin_amdgcn_wmma_f32_16x16x32_bf16(
                        false, afrag[mt], false, bfrag[nt],
                        (short)0, acc[mt][nt], false, false);
        }
        __syncthreads();   // all waves done reading buf[cur] before it is restaged
    }

    // ---- epilogue: C/D layout -> VGPR r holds row (r + 8*halfSel) of the tile
    #pragma unroll
    for (int mt = 0; mt < 2; ++mt) {
        const int rowB = mBase + mt * 16 + (halfSel << 3);   // 8 consecutive rows
        #pragma unroll
        for (int nt = 0; nt < 2; ++nt) {
            const int col = nBase + wave * 32 + nt * 16 + l15;
            float v[8];
            #pragma unroll
            for (int r = 0; r < 8; ++r) {
                v[r] = acc[mt][nt][r];
                if (Cadd)
                    v[r] += (float)Cadd[(long)batch * cBatch + (long)(rowB + r) * cRow + col];
            }
            if (OutF) {
                #pragma unroll
                for (int r = 0; r < 8; ++r)
                    OutF[(long)batch * oBatch + (long)(rowB + r) * oRow + col] = v[r];
            }
            if (OutB) {
                #pragma unroll
                for (int r = 0; r < 8; ++r)
                    OutB[(long)batch * oBatch + (long)(rowB + r) * oRow + col] = (__bf16)v[r];
            }
            if (OutT) {  // transposed store: 8 consecutive rows -> one 16B store
                __bf16 t[8];
                #pragma unroll
                for (int r = 0; r < 8; ++r) t[r] = (__bf16)v[r];
                *(uint4*)&OutT[(long)col * Mtot + rowB] = *(const uint4*)t;
            }
        }
    }
}

// ---- host ------------------------------------------------------------------

extern "C" void kernel_launch(void* const* d_in, const int* in_sizes, int n_in,
                              void* d_out, int out_size, void* d_ws, size_t ws_size,
                              hipStream_t stream) {
    const int B = 32, T = 2048, U = 1024, D = 1024;

    const float* x  = (const float*)d_in[0];
    const float* Wk = (const float*)d_in[2];    // [D, U]
    const float* Wr = (const float*)d_in[3];    // [U, U]
    float*       out = (float*)d_out;           // [B, U] f32

    char* ws = (char*)d_ws;
    const size_t NX = (size_t)B * T * D;        // 64Mi elems
    __bf16* xbf  = (__bf16*)ws;                                  // 128 MB (dead after phase 1)
    __bf16* xinb = (__bf16*)(ws + NX * 2);                       // 128 MB
    __bf16* bufP = (__bf16*)(ws + NX * 4);                       // 64 MB
    size_t o = NX * 4 + (size_t)(T / 2) * B * U * 2;
    __bf16* WkT = (__bf16*)(ws + o);  o += (size_t)D * U * 2;
    __bf16 *P[3], *PT[3];
    for (int i = 0; i < 3; ++i) {
        P[i]  = (__bf16*)(ws + o); o += (size_t)U * U * 2;
        PT[i] = (__bf16*)(ws + o); o += (size_t)U * U * 2;
    }

    dim3 blk(256, 1, 1);

    // one-shot converts (bf16 operands; B-operands pre-transposed to N-major)
    cvt_f32_bf16  <<<dim3((unsigned)(NX / 2048)), blk, 0, stream>>>(x,  xbf,  (long)NX);
    cvt_f32_bf16  <<<dim3(U * U / 2048),          blk, 0, stream>>>(Wr, P[0], (long)U * U);
    cvt_f32_bf16_T<<<dim3(D * U / 256),           blk, 0, stream>>>(Wk, WkT,  D, U);
    cvt_f32_bf16_T<<<dim3(U * U / 256),           blk, 0, stream>>>(Wr, PT[0], U, U);

    // Phase 1: xin = x @ Wk   ([B*T, D] @ [D, U]) -> bf16
    minrnn_gemm<<<dim3(U / TILE_N, (B * T) / TILE_M, 1), blk, 0, stream>>>(
        xbf, (long)D, 0L, WkT,
        nullptr, 0L, 0L,
        xinb, nullptr, (long)U, 0L,
        nullptr, 0, U, D);

    // Phase 2: log-depth linear scan. b_i = a_{2i} @ P + a_{2i+1};  P <- P^2.
    const __bf16* Pc  = P[0];
    const __bf16* PTc = PT[0];
    const __bf16* inBase = xinb;
    long inRow = (long)T * U, inItem = (long)U;
    int Lcur = T, nxt = 1;

    for (int lvl = 0; lvl < 11; ++lvl) {
        const int pairs = Lcur / 2;
        const bool last = (pairs == 1);
        __bf16* outB = last ? nullptr : ((lvl & 1) ? xbf : bufP);
        float*  outF = last ? out : nullptr;

        minrnn_gemm<<<dim3(U / TILE_N, 1, pairs), blk, 0, stream>>>(
            inBase, inRow, 2L * inItem, PTc,
            inBase + inItem, inRow, 2L * inItem,
            outB, outF, (long)U, (long)B * U,
            nullptr, 0, U, U);

        if (!last) {  // P <- P @ P, emitting both P and P^T for the next level
            minrnn_gemm<<<dim3(U / TILE_N, U / TILE_M, 1), blk, 0, stream>>>(
                Pc, (long)U, 0L, PTc,
                nullptr, 0L, 0L,
                P[nxt], nullptr, (long)U, 0L,
                PT[nxt], U, U, U);
            Pc = P[nxt]; PTc = PT[nxt];
            nxt = (nxt == 1) ? 2 : 1;
        }

        inBase = outB; inRow = (long)U; inItem = (long)B * U; Lcur = pairs;
    }
}
